// MultiHeadedAttentionAug_35940286333504
// MI455X (gfx1250) — matrix operand order repaired and checked by reading, not compiled, not run on hardware
//
#include <hip/hip_runtime.h>
#include <hip/hip_bf16.h>
#include <cstdint>

// ---------------------------------------------------------------------------
// MultiHeadedAttentionAug for MI455X (gfx1250, wave32, WMMA bf16 path)
// B=8, S=1024, D=512, H=8, DK=64
//
// Pipeline:
//  1) gemm512_bias x3 : q/k/v projections [B*S,512]@[512,512]^T -> [B,H,S,64]
//  2) refine_kernel   : 64x64 linears + elu/relu/softmax -> asp,q2,kr,vr (bf16)
//                       plus scalars q1m[s], cwk[t]=cw.kr[t], cbk[t]=cb.kr[t]
//                       (alpha_channel is rank-2 => ac[s,t]=q1m[s]*cwk[t]+cbk[t])
//  3) aug_attn        : flash-style fused (s,t) tiling, all GEMMs via
//                       v_wmma_f32_16x16x32_bf16; bf16 tiles staged with
//                       global_load_async_to_lds_b128 (ASYNCcnt path)
//  4) gemm512_bias    : output projection -> d_out
//
// Workspace: ~102 MB (see layout in kernel_launch).
// ---------------------------------------------------------------------------

#define BB 8
#define SS 1024
#define DD 512
#define HH 8
#define DK 64
#define LDP 72   // LDS row stride in bf16 elems (144 B = 9*16 B, 16B aligned)

typedef __bf16 bf16_t;
typedef __attribute__((ext_vector_type(16))) __bf16 v16bf;
typedef __attribute__((ext_vector_type(8)))  __bf16 v8bf;
typedef __attribute__((ext_vector_type(8)))  float  v8f;

__device__ __forceinline__ bf16_t f2bf(float f) {
  unsigned int u = __builtin_bit_cast(unsigned int, f);
  unsigned int r = (u + 0x7FFFu + ((u >> 16) & 1u)) >> 16;   // round-nearest-even
  return __builtin_bit_cast(bf16_t, (unsigned short)r);
}

__device__ __forceinline__ float eluf(float x) {
  return x > 0.f ? x : (__expf(x) - 1.f);
}

__device__ __forceinline__ v8f wmma_bf16(v16bf a, v16bf b, v8f c) {
  return __builtin_amdgcn_wmma_f32_16x16x32_bf16(
      /*neg_a=*/false, a, /*neg_b=*/false, b,
      /*c_mod=*/(short)0, c, /*reuse_a=*/false, /*reuse_b=*/false);
}

// Async global->LDS 16-byte copy (CDNA5 ASYNCcnt path). vdst = wave-relative
// LDS byte address (low 32 bits of the generic shared pointer, per flat-LDS
// aperture truncation), vaddr = 64-bit global address.
__device__ __forceinline__ void async_copy_b128(void* lds_ptr, const void* gptr) {
  unsigned lds_off = (unsigned)(uintptr_t)lds_ptr;
  unsigned long long ga = (unsigned long long)(uintptr_t)gptr;
  asm volatile("global_load_async_to_lds_b128 %0, %1, off"
               :: "v"(lds_off), "v"(ga)
               : "memory");
}

__device__ __forceinline__ void wait_async0() {
  asm volatile("s_wait_asynccnt 0x0" ::: "memory");
}

// A-matrix fragment (16x32 bf16, MxK) from LDS stored row-major [row][k], stride LDP.
// ISA layout: lane r=lane&15 -> row; half=lane>>4 -> K chunks {8h..8h+7, 16+8h..16+8h+7}
__device__ __forceinline__ v16bf load_frag_a(const bf16_t* base, int row0, int k0) {
  int lane = threadIdx.x & 31;
  int r = lane & 15, hf = lane >> 4;
  const bf16_t* p = base + (row0 + r) * LDP + k0 + hf * 8;
  v8bf lo = *(const v8bf*)p;
  v8bf hi = *(const v8bf*)(p + 16);
  v16bf o;
#pragma unroll
  for (int i = 0; i < 8; i++) { o[i] = lo[i]; o[8 + i] = hi[i]; }
  return o;
}

// B-matrix fragment (32x16 bf16, KxN) from LDS stored as [n][k] row-major, stride LDP.
// ISA layout: lane n=lane&15 -> col; half=lane>>4 -> K chunks {8h..8h+7, 16+8h..16+8h+7}
__device__ __forceinline__ v16bf load_frag_b(const bf16_t* base, int col0, int k0) {
  int lane = threadIdx.x & 31;
  int n = lane & 15, hf = lane >> 4;
  const bf16_t* p = base + (col0 + n) * LDP + k0 + hf * 8;
  v8bf lo = *(const v8bf*)p;
  v8bf hi = *(const v8bf*)(p + 16);
  v16bf o;
#pragma unroll
  for (int i = 0; i < 8; i++) { o[i] = lo[i]; o[8 + i] = hi[i]; }
  return o;
}

// ---------------------------------------------------------------------------
// Generic 512-K GEMM: Y = X @ W^T + bias.  X:[R,512] f32, W:[512,512] f32.
// permute==0: Y[r*512+n]; permute==1: Y laid out as [B,H,S,64] (n = h*64+dk).
// Block 128 thr (4 waves): 64x64 tile; wave w owns 16-row strip (4 C tiles).
// ---------------------------------------------------------------------------
__global__ __launch_bounds__(128) void gemm512_bias(
    const float* __restrict__ X, const float* __restrict__ W,
    const float* __restrict__ bias, float* __restrict__ Y, int permute) {
  __shared__ __align__(16) bf16_t Xs[64 * LDP];
  __shared__ __align__(16) bf16_t Ws[64 * LDP];
  const int tid = threadIdx.x;
  const int wave = tid >> 5, lane = tid & 31;
  const int half = lane >> 4, nl = lane & 15;
  const int r0 = blockIdx.x * 64;
  const int n0 = blockIdx.y * 64;

  v8f acc[4] = {};

  for (int k0 = 0; k0 < 512; k0 += 64) {
    // cooperative load + f32->bf16 convert: 64x64 X tile and 64x64 W tile
    const int row = tid >> 1, hf = tid & 1;
    const float4* gx = (const float4*)(X + (size_t)(r0 + row) * 512 + k0 + hf * 32);
    const float4* gw = (const float4*)(W + (size_t)(n0 + row) * 512 + k0 + hf * 32);
    bf16_t* dx = Xs + row * LDP + hf * 32;
    bf16_t* dw = Ws + row * LDP + hf * 32;
    if (k0 + 64 < 512) {  // prefetch next k-slab (global_prefetch_b8)
      __builtin_prefetch(X + (size_t)(r0 + row) * 512 + k0 + 64 + hf * 32, 0, 0);
      __builtin_prefetch(W + (size_t)(n0 + row) * 512 + k0 + 64 + hf * 32, 0, 0);
    }
#pragma unroll
    for (int i = 0; i < 8; i++) {
      float4 a = gx[i];
      dx[i * 4 + 0] = f2bf(a.x); dx[i * 4 + 1] = f2bf(a.y);
      dx[i * 4 + 2] = f2bf(a.z); dx[i * 4 + 3] = f2bf(a.w);
      float4 b = gw[i];
      dw[i * 4 + 0] = f2bf(b.x); dw[i * 4 + 1] = f2bf(b.y);
      dw[i * 4 + 2] = f2bf(b.z); dw[i * 4 + 3] = f2bf(b.w);
    }
    __syncthreads();

    v16bf a0 = load_frag_a(Xs, wave * 16, 0);
    v16bf a1 = load_frag_a(Xs, wave * 16, 32);
#pragma unroll
    for (int tn = 0; tn < 4; tn++) {
      v16bf b0 = load_frag_b(Ws, tn * 16, 0);
      v16bf b1 = load_frag_b(Ws, tn * 16, 32);
      acc[tn] = wmma_bf16(a0, b0, acc[tn]);
      acc[tn] = wmma_bf16(a1, b1, acc[tn]);
    }
    __syncthreads();
  }

  // epilogue: bias + store (C/D layout: lane= N, vgpr j = M row, half => M+8)
#pragma unroll
  for (int tn = 0; tn < 4; tn++) {
    const int n = n0 + tn * 16 + nl;
    const float bv = bias[n];
#pragma unroll
    for (int j = 0; j < 8; j++) {
      const int m = r0 + wave * 16 + (half ? 8 + j : j);
      const float val = acc[tn][j] + bv;
      size_t idx;
      if (permute) {
        const int b = m >> 10, s = m & 1023;        // S = 1024
        const int h = n >> 6, dk = n & 63;
        idx = ((((size_t)b * HH + h) * SS) + s) * DK + dk;
      } else {
        idx = (size_t)m * 512 + n;
      }
      Y[idx] = val;
    }
  }
}

// ---------------------------------------------------------------------------
// Refine path: per flattened row r in [B*H*S), 64x64 matvecs + activations.
// Produces bf16 asp/q2/kr/vr and f32 q1m/cwk/cbk.
// ---------------------------------------------------------------------------
__global__ __launch_bounds__(64) void refine_kernel(
    const float* __restrict__ qp, const float* __restrict__ kp,
    const float* __restrict__ vp,
    const float* __restrict__ query_w, const float* __restrict__ query_b,
    const float* __restrict__ embed_w, const float* __restrict__ embed_b,
    const float* __restrict__ spatial_w, const float* __restrict__ spatial_b,
    const float* __restrict__ query2_w, const float* __restrict__ query2_b,
    const float* __restrict__ key_w, const float* __restrict__ key_b,
    const float* __restrict__ value_w, const float* __restrict__ value_b,
    const float* __restrict__ channel_w, const float* __restrict__ channel_b,
    bf16_t* __restrict__ asp_o, bf16_t* __restrict__ q2_o,
    bf16_t* __restrict__ kr_o, bf16_t* __restrict__ vr_o,
    float* __restrict__ q1m_o, float* __restrict__ cwk_o,
    float* __restrict__ cbk_o) {
  __shared__ float sQ[64], sK[64], sV[64], sT[64], sQ1[64], sSP[64], sKR[64];
  const size_t row = blockIdx.x;
  const int n = threadIdx.x;
  const size_t base = row * 64;

  sQ[n] = qp[base + n];
  sK[n] = kp[base + n];
  sV[n] = vp[base + n];
  __syncthreads();

  // t1 = elu(Q @ query_w^T + query_b)
  float acc = query_b[n];
#pragma unroll 8
  for (int d = 0; d < 64; d++) acc += sQ[d] * query_w[n * 64 + d];
  sT[n] = eluf(acc);
  __syncthreads();

  // q1 = relu(t1 @ embed_w^T + embed_b)
  acc = embed_b[n];
#pragma unroll 8
  for (int d = 0; d < 64; d++) acc += sT[d] * embed_w[n * 64 + d];
  sQ1[n] = fmaxf(acc, 0.f);
  __syncthreads();

  // sp = q1 @ spatial_w^T + spatial_b ; asp = softmax(sp)
  acc = spatial_b[n];
#pragma unroll 8
  for (int d = 0; d < 64; d++) acc += sQ1[d] * spatial_w[n * 64 + d];
  sSP[n] = acc;
  __syncthreads();
  float mx = -3.4e38f;
  for (int d = 0; d < 64; d++) mx = fmaxf(mx, sSP[d]);
  float sum = 0.f;
  for (int d = 0; d < 64; d++) sum += __expf(sSP[d] - mx);
  asp_o[base + n] = f2bf(__expf(sSP[n] - mx) / sum);

  if (n == 0) {  // q1m = mean(q1)
    float s = 0.f;
    for (int d = 0; d < 64; d++) s += sQ1[d];
    q1m_o[row] = s * (1.f / 64.f);
  }

  // q2 = elu(Q @ query2_w^T + query2_b)
  acc = query2_b[n];
#pragma unroll 8
  for (int d = 0; d < 64; d++) acc += sQ[d] * query2_w[n * 64 + d];
  q2_o[base + n] = f2bf(eluf(acc));

  // kr = elu(K @ key_w^T + key_b)
  acc = key_b[n];
#pragma unroll 8
  for (int d = 0; d < 64; d++) acc += sK[d] * key_w[n * 64 + d];
  float kr = eluf(acc);
  sKR[n] = kr;
  kr_o[base + n] = f2bf(kr);
  __syncthreads();
  if (n == 0) {  // cwk = channel_w . kr   (channel_w is [64,1])
    float s = 0.f;
    for (int d = 0; d < 64; d++) s += channel_w[d] * sKR[d];
    cwk_o[row] = s;
  }
  if (n == 1) {  // cbk = channel_b . kr
    float s = 0.f;
    for (int d = 0; d < 64; d++) s += channel_b[d] * sKR[d];
    cbk_o[row] = s;
  }

  // vr = elu(V @ value_w^T + value_b)
  acc = value_b[n];
#pragma unroll 8
  for (int d = 0; d < 64; d++) acc += sV[d] * value_w[n * 64 + d];
  vr_o[base + n] = f2bf(eluf(acc));
}

// ---------------------------------------------------------------------------
// Fused augmented-attention core. Block = 128 thr (4 waves), owns a 64-row
// s-block of one (b,h) head. Loops over t in 64-wide tiles:
//   S1 = asp @ kr^T ; S2 = q2 @ vr^T                (WMMA bf16)
//   W  = (q1m[s]*cwk[t] + cbk[t]) * S1 * S2         (registers, rank-2 ac)
//   out += W @ v                                    (WMMA bf16 via LDS stage)
// bf16 tiles staged via global_load_async_to_lds_b128 (ASYNCcnt), v tile
// converted/transposed on the VALU concurrently; next t-tile prefetched.
// ---------------------------------------------------------------------------
__global__ __launch_bounds__(128) void aug_attn(
    const float* __restrict__ vp,                  // [B,H,S,64] f32
    const bf16_t* __restrict__ asp, const bf16_t* __restrict__ q2,
    const bf16_t* __restrict__ kr, const bf16_t* __restrict__ vr,
    const float* __restrict__ q1m, const float* __restrict__ cwk,
    const float* __restrict__ cbk,
    float* __restrict__ mid) {                     // [B,S,512] f32
  __shared__ __align__(16) bf16_t sASP[64 * LDP];
  __shared__ __align__(16) bf16_t sQ2[64 * LDP];
  __shared__ __align__(16) bf16_t sKR[64 * LDP];
  __shared__ __align__(16) bf16_t sVR[64 * LDP];
  __shared__ __align__(16) bf16_t sVT[64 * LDP];   // v transposed: [d][t]
  __shared__ __align__(16) bf16_t sW[64 * LDP];    // staged W tile (bf16)
  __shared__ float sQ1M[64], sCW[64], sCB[64];

  const int tid = threadIdx.x;
  const int wave = tid >> 5, lane = tid & 31;
  const int half = lane >> 4, nl = lane & 15;
  const int s0 = blockIdx.x * 64;
  const int bh = blockIdx.y;                 // b*H + h
  const int b = bh >> 3, h = bh & 7;
  const size_t headbase = (size_t)bh * SS * 64;

  const int row = tid >> 1, hf = tid & 1;

  // persistent per-s tiles: asp, q2 (bf16, async copy), q1m
  {
    const bf16_t* ga = asp + headbase + (size_t)(s0 + row) * 64 + hf * 32;
    const bf16_t* gq = q2 + headbase + (size_t)(s0 + row) * 64 + hf * 32;
    bf16_t* da = sASP + row * LDP + hf * 32;
    bf16_t* dq = sQ2 + row * LDP + hf * 32;
#pragma unroll
    for (int i = 0; i < 4; i++) {
      async_copy_b128(da + i * 8, ga + i * 8);
      async_copy_b128(dq + i * 8, gq + i * 8);
    }
    if (tid < 64) sQ1M[tid] = q1m[(size_t)bh * SS + s0 + tid];
  }

  v8f acc[4] = {};

  for (int t0 = 0; t0 < SS; t0 += 64) {
    // per-t tiles: kr, vr (bf16, async), v (f32 -> bf16 transposed), cwk/cbk
    {
      const bf16_t* gk = kr + headbase + (size_t)(t0 + row) * 64 + hf * 32;
      const bf16_t* gv = vr + headbase + (size_t)(t0 + row) * 64 + hf * 32;
      bf16_t* dk = sKR + row * LDP + hf * 32;
      bf16_t* dv = sVR + row * LDP + hf * 32;
#pragma unroll
      for (int i = 0; i < 4; i++) {
        async_copy_b128(dk + i * 8, gk + i * 8);
        async_copy_b128(dv + i * 8, gv + i * 8);
      }
      if (t0 + 64 < SS) {  // prefetch next t-tile while WMMAs run
        __builtin_prefetch(vp + headbase + (size_t)(t0 + 64 + row) * 64 + hf * 32, 0, 0);
        __builtin_prefetch(kr + headbase + (size_t)(t0 + 64 + row) * 64 + hf * 32, 0, 0);
        __builtin_prefetch(vr + headbase + (size_t)(t0 + 64 + row) * 64 + hf * 32, 0, 0);
      }
      const float4* gvv = (const float4*)(vp + headbase + (size_t)(t0 + row) * 64 + hf * 32);
#pragma unroll
      for (int i = 0; i < 8; i++) {
        float4 x = gvv[i];
        const int d = hf * 32 + i * 4;
        sVT[(d + 0) * LDP + row] = f2bf(x.x);
        sVT[(d + 1) * LDP + row] = f2bf(x.y);
        sVT[(d + 2) * LDP + row] = f2bf(x.z);
        sVT[(d + 3) * LDP + row] = f2bf(x.w);
      }
      if (tid < 64) sCW[tid] = cwk[(size_t)bh * SS + t0 + tid];
      else          sCB[tid - 64] = cbk[(size_t)bh * SS + t0 + tid - 64];
    }
    wait_async0();     // our async tile copies have landed in LDS
    __syncthreads();

    // S1 = asp @ kr^T ; S2 = q2 @ vr^T ; combine with rank-2 ac term
    v16bf a0 = load_frag_a(sASP, wave * 16, 0);
    v16bf a1 = load_frag_a(sASP, wave * 16, 32);
    v16bf c0 = load_frag_a(sQ2, wave * 16, 0);
    v16bf c1 = load_frag_a(sQ2, wave * 16, 32);
    v8f zero = {};
#pragma unroll
    for (int tn = 0; tn < 4; tn++) {
      v16bf b0 = load_frag_b(sKR, tn * 16, 0);
      v16bf b1 = load_frag_b(sKR, tn * 16, 32);
      v8f s1 = wmma_bf16(a0, b0, zero);
      s1 = wmma_bf16(a1, b1, s1);
      v16bf d0 = load_frag_b(sVR, tn * 16, 0);
      v16bf d1 = load_frag_b(sVR, tn * 16, 32);
      v8f s2 = wmma_bf16(c0, d0, zero);
      s2 = wmma_bf16(c1, d1, s2);
      const int tl = tn * 16 + nl;
      const float cw = sCW[tl], cb = sCB[tl];
#pragma unroll
      for (int j = 0; j < 8; j++) {
        const int mm = wave * 16 + (half ? 8 + j : j);
        const float wv = (sQ1M[mm] * cw + cb) * s1[j] * s2[j];
        sW[mm * LDP + tl] = f2bf(wv);
      }
    }
    __syncthreads();

    // out += W @ v   (A = sW strip over t, B = sVT [d][t])
    v16bf w0 = load_frag_a(sW, wave * 16, 0);
    v16bf w1 = load_frag_a(sW, wave * 16, 32);
#pragma unroll
    for (int dn = 0; dn < 4; dn++) {
      v16bf b0 = load_frag_b(sVT, dn * 16, 0);
      v16bf b1 = load_frag_b(sVT, dn * 16, 32);
      acc[dn] = wmma_bf16(w0, b0, acc[dn]);
      acc[dn] = wmma_bf16(w1, b1, acc[dn]);
    }
    __syncthreads();
  }

  // store strip to mid[b, s, h*64 + d]
#pragma unroll
  for (int dn = 0; dn < 4; dn++) {
    const int d = dn * 16 + nl;
#pragma unroll
    for (int j = 0; j < 8; j++) {
      const int mm = wave * 16 + (half ? 8 + j : j);
      mid[(((size_t)b * SS + s0 + mm) * 512) + h * 64 + d] = acc[dn][j];
    }
  }
}

// ---------------------------------------------------------------------------
// Launch. Input order (setup_inputs dict order, params nested in order):
//  0 query  1 key  2 value
//  3 q_w 4 q_b 5 k_w 6 k_b 7 v_w 8 v_b 9 o_w 10 o_b
// 11 key_w 12 key_b 13 query_w 14 query_b 15 query2_w 16 query2_b
// 17 value_w 18 value_b 19 embed_w 20 embed_b 21 spatial_w 22 spatial_b
// 23 channel_w 24 channel_b
// ---------------------------------------------------------------------------
extern "C" void kernel_launch(void* const* d_in, const int* in_sizes, int n_in,
                              void* d_out, int out_size, void* d_ws, size_t ws_size,
                              hipStream_t stream) {
  const float* query = (const float*)d_in[0];
  const float* key = (const float*)d_in[1];
  const float* value = (const float*)d_in[2];
  const float* q_w = (const float*)d_in[3];
  const float* q_b = (const float*)d_in[4];
  const float* k_w = (const float*)d_in[5];
  const float* k_b = (const float*)d_in[6];
  const float* v_w = (const float*)d_in[7];
  const float* v_b = (const float*)d_in[8];
  const float* o_w = (const float*)d_in[9];
  const float* o_b = (const float*)d_in[10];
  const float* key_w = (const float*)d_in[11];
  const float* key_b = (const float*)d_in[12];
  const float* query_w = (const float*)d_in[13];
  const float* query_b = (const float*)d_in[14];
  const float* query2_w = (const float*)d_in[15];
  const float* query2_b = (const float*)d_in[16];
  const float* value_w = (const float*)d_in[17];
  const float* value_b = (const float*)d_in[18];
  const float* embed_w = (const float*)d_in[19];
  const float* embed_b = (const float*)d_in[20];
  const float* spatial_w = (const float*)d_in[21];
  const float* spatial_b = (const float*)d_in[22];
  const float* channel_w = (const float*)d_in[23];
  const float* channel_b = (const float*)d_in[24];

  // workspace layout (needs ~102 MB)
  const size_t NE = (size_t)BB * HH * SS * DK;  // 4,194,304
  float* qp = (float*)d_ws;
  float* kp = qp + NE;
  float* vp = kp + NE;
  float* mid = vp + NE;                      // [B,S,512]
  bf16_t* aspb = (bf16_t*)(mid + NE);
  bf16_t* q2b = aspb + NE;
  bf16_t* krb = q2b + NE;
  bf16_t* vrb = krb + NE;
  float* q1mw = (float*)(vrb + NE);
  float* cwkw = q1mw + (size_t)BB * HH * SS;
  float* cbkw = cwkw + (size_t)BB * HH * SS;

  const dim3 gproj(BB * SS / 64, DD / 64);   // (128, 8)

  // 1) projections -> [B,H,S,64]
  gemm512_bias<<<gproj, 128, 0, stream>>>(query, q_w, q_b, qp, 1);
  gemm512_bias<<<gproj, 128, 0, stream>>>(key, k_w, k_b, kp, 1);
  gemm512_bias<<<gproj, 128, 0, stream>>>(value, v_w, v_b, vp, 1);

  // 2) refine path
  refine_kernel<<<(unsigned)(BB * HH * SS), 64, 0, stream>>>(
      qp, kp, vp, query_w, query_b, embed_w, embed_b, spatial_w, spatial_b,
      query2_w, query2_b, key_w, key_b, value_w, value_b, channel_w, channel_b,
      aspb, q2b, krb, vrb, q1mw, cwkw, cbkw);

  // 3) fused attention core -> mid [B,S,512]
  aug_attn<<<dim3(SS / 64, BB * HH), 128, 0, stream>>>(
      vp, aspb, q2b, krb, vrb, q1mw, cwkw, cbkw, mid);

  // 4) output projection -> d_out [B,S,512]
  gemm512_bias<<<gproj, 128, 0, stream>>>(mid, o_w, o_b, (float*)d_out, 0);
}
